// WindowAttention_17016660426858
// MI455X (gfx1250) — matrix operand order repaired and checked by reading, not compiled, not run on hardware
//
#include <hip/hip_runtime.h>
#include <hip/hip_bf16.h>

typedef __attribute__((ext_vector_type(16))) _Float16 v16h;
typedef __attribute__((ext_vector_type(8)))  _Float16 v8h;
typedef __attribute__((ext_vector_type(8)))  float    v8f;

#define HEADS 6
#define HD    32
#define NTOK  49
#define DIM   192
#define NWIN  64
#define BTOT  4096
#define MROWS (BTOT * NTOK) /* 200704 */

union V16 { v16h v; v8h h[2]; };

// A-fragment (16x32 f16): lane m = lane&15, half = lane>>4.
// a[e] = row[(e>>3)*16 + half*8 + (e&7)] -> two contiguous 8-halfword runs.
__device__ __forceinline__ v16h ldsA(const _Float16* rowBase, int half) {
    V16 u;
    u.h[0] = *(const v8h*)(rowBase + half * 8);
    u.h[1] = *(const v8h*)(rowBase + 16 + half * 8);
    return u.v;
}
// B-fragment (32x16 f16) from [n][k] storage: lane n = lane&15, half = lane>>4.
// b[e] = rowN[(e>>3)*16 + half*8 + (e&7)] -> two contiguous 8-halfword runs.
__device__ __forceinline__ v16h ldsB(const _Float16* rowN, int half) {
    V16 u;
    u.h[0] = *(const v8h*)(rowN + half * 8);
    u.h[1] = *(const v8h*)(rowN + 16 + half * 8);
    return u.v;
}

// ---------------------------------------------------------------------------
// Kernel 1: QKV projection. Per block: C[64 x 96] = X[64 x 192] * W^T + b.
// Whole A strip and B panel staged to LDS once; barrier-free WMMA k-loop.
// grid (6, MROWS/64)  [x = column panel -> adjacent blocks share A strip in L2]
// block 256 (8 waves).
// ---------------------------------------------------------------------------
__global__ __launch_bounds__(256) void qkv_gemm(
    const float* __restrict__ x, const float* __restrict__ qkv_w,
    const float* __restrict__ qkv_b,
    _Float16* __restrict__ qf, _Float16* __restrict__ kf,
    _Float16* __restrict__ vf)
{
    __shared__ _Float16 As[64][200];  // [m][k], k padded 192+8
    __shared__ _Float16 Bs[96][200];  // [n][k]

    const int tid = threadIdx.x;
    const int wave = tid >> 5, lane = tid & 31;
    const int lane15 = lane & 15, half = lane >> 4;
    const int n0 = blockIdx.x * 96;   // 0..480
    const int m0 = blockIdx.y * 64;
    const int mt = wave >> 1;         // 0..3
    const int ntb = (wave & 1) * 3;   // 3 n-tiles per wave

    // stage A: 64 x 192 fp32 -> f16 (12 float4 per thread, one row each 4 thr)
    {
        const float4* srow =
            (const float4*)x + ((size_t)(m0 + (tid >> 2)) * DIM) / 4;
        _Float16* drow = &As[tid >> 2][0];
        const int cbase = (tid & 3) * 12;
#pragma unroll
        for (int j = 0; j < 12; ++j) {
            float4 vv = srow[cbase + j];
            _Float16* dst = drow + (cbase + j) * 4;
            dst[0] = (_Float16)vv.x; dst[1] = (_Float16)vv.y;
            dst[2] = (_Float16)vv.z; dst[3] = (_Float16)vv.w;
        }
    }
    // stage B: 96 x 192 fp32 -> f16
    {
        const float4* src = (const float4*)qkv_w;
        for (int i = tid; i < 96 * 48; i += 256) {
            int n = i / 48, c4 = i - n * 48;
            float4 vv = src[((size_t)(n0 + n) * DIM) / 4 + c4];
            _Float16* dst = &Bs[n][c4 * 4];
            dst[0] = (_Float16)vv.x; dst[1] = (_Float16)vv.y;
            dst[2] = (_Float16)vv.z; dst[3] = (_Float16)vv.w;
        }
    }
    __syncthreads();

    v8f acc[3];
#pragma unroll
    for (int t = 0; t < 3; ++t) acc[t] = (v8f){};

#pragma unroll
    for (int kk = 0; kk < 6; ++kk) {
        v16h a = ldsA(&As[mt * 16 + lane15][kk * 32], half);
#pragma unroll
        for (int t = 0; t < 3; ++t) {
            v16h b = ldsB(&Bs[(ntb + t) * 16 + lane15][kk * 32], half);
            acc[t] = __builtin_amdgcn_wmma_f32_16x16x32_f16(
                false, a, false, b, (short)0, acc[t], false, false);
        }
    }

    const float scale = 0.17677669529663687f; // 32^-0.5
#pragma unroll
    for (int t = 0; t < 3; ++t) {
        int col = n0 + (ntb + t) * 16 + lane15;  // 0..575
        int which = col / DIM;                   // 0=q 1=k 2=v
        int rem = col - which * DIM;
        int h = rem >> 5, d = rem & 31;
        float bias = qkv_b[col];
        _Float16* dst = (which == 0) ? qf : (which == 1) ? kf : vf;
        float mul = (which == 0) ? scale : 1.0f;
#pragma unroll
        for (int r = 0; r < 8; ++r) {
            int gm = m0 + mt * 16 + half * 8 + r;
            int bwin = gm / NTOK;
            int nrow = gm - bwin * NTOK;
            float val = (acc[t][r] + bias) * mul;
            dst[(((size_t)bwin * HEADS + h) * NTOK + nrow) * HD + d] =
                (_Float16)val;
        }
    }
}

// ---------------------------------------------------------------------------
// Kernel 2: windowed attention for one (window b, head h) per block.
// grid BTOT*HEADS, block 128 (4 waves); tokens padded 49 -> 64.
// ---------------------------------------------------------------------------
__global__ __launch_bounds__(128) void win_attn(
    const _Float16* __restrict__ qf, const _Float16* __restrict__ kf,
    const _Float16* __restrict__ vf, const float* __restrict__ mask,
    const float* __restrict__ bias_table, const int* __restrict__ rel_index,
    _Float16* __restrict__ aout)
{
    __shared__ _Float16 Ql[64][40];   // [token][d]
    __shared__ _Float16 Kl[64][40];   // [token][d]
    __shared__ _Float16 VlT[32][72];  // [d][token]
    __shared__ _Float16 Pl[64][72];   // [m][j] softmax probs

    const int tid = threadIdx.x;
    const int wave = tid >> 5, lane = tid & 31;
    const int lane15 = lane & 15, half = lane >> 4;
    const int bh = blockIdx.x;
    const int b = bh / HEADS, h = bh - b * HEADS;
    const int w = b & (NWIN - 1);

    const size_t base = (size_t)bh * NTOK * HD;   // 16B-aligned (3136 B rows)
    const uint4* qg4 = (const uint4*)(qf + base);
    const uint4* kg4 = (const uint4*)(kf + base);
    const unsigned* vg = (const unsigned*)(vf + base);

    // Q, K: b128 copies, zero-padded rows 49..63
    for (int i = tid; i < 64 * 4; i += 128) {
        int row = i >> 2, c4 = i & 3;
        uint4 qv = {0u, 0u, 0u, 0u}, kv = {0u, 0u, 0u, 0u};
        if (row < NTOK) {
            qv = qg4[row * 4 + c4];
            kv = kg4[row * 4 + c4];
        }
        *(uint4*)&Ql[row][c4 * 8] = qv;
        *(uint4*)&Kl[row][c4 * 8] = kv;
    }
    // V: transpose into [d][token]
    for (int i = tid; i < 64 * 16; i += 128) {
        int row = i >> 4, c = i & 15;
        unsigned vv = 0;
        if (row < NTOK) vv = vg[row * 16 + c];
        union { unsigned u; _Float16 f[2]; } uv; uv.u = vv;
        VlT[c * 2 + 0][row] = uv.f[0];
        VlT[c * 2 + 1][row] = uv.f[1];
    }
    __syncthreads();

    // S = Q @ K^T : wave owns m-tile = wave, all 4 n-tiles. K=hd=32 -> 1 WMMA.
    const int mt = wave;
    v8f s[4];
    {
        v16h a = ldsA(&Ql[mt * 16 + lane15][0], half);
#pragma unroll
        for (int nt = 0; nt < 4; ++nt) {
            v8f z = {};
            v16h bfr = ldsB(&Kl[nt * 16 + lane15][0], half);
            s[nt] = __builtin_amdgcn_wmma_f32_16x16x32_f16(
                false, a, false, bfr, (short)0, z, false, false);
        }
    }

    // bias + mask, -inf padding
#pragma unroll
    for (int nt = 0; nt < 4; ++nt) {
        int n = nt * 16 + lane15;
#pragma unroll
        for (int r = 0; r < 8; ++r) {
            int m = mt * 16 + half * 8 + r;
            float v = s[nt][r];
            if (m < NTOK && n < NTOK) {
                int rel = rel_index[m * NTOK + n];
                v += bias_table[rel * HEADS + h] +
                     mask[((size_t)w * NTOK + m) * NTOK + n];
            } else {
                v = -1e30f;
            }
            s[nt][r] = v;
        }
    }

    // row softmax: row m lives in the 16 lanes of one half (xor 1,2,4,8)
    float rmax[8], inv[8];
#pragma unroll
    for (int r = 0; r < 8; ++r) {
        float mx = fmaxf(fmaxf(s[0][r], s[1][r]), fmaxf(s[2][r], s[3][r]));
        mx = fmaxf(mx, __shfl_xor(mx, 1, 32));
        mx = fmaxf(mx, __shfl_xor(mx, 2, 32));
        mx = fmaxf(mx, __shfl_xor(mx, 4, 32));
        mx = fmaxf(mx, __shfl_xor(mx, 8, 32));
        rmax[r] = mx;
    }
#pragma unroll
    for (int r = 0; r < 8; ++r) {
        float sum = 0.f;
#pragma unroll
        for (int nt = 0; nt < 4; ++nt) {
            float e = __expf(s[nt][r] - rmax[r]);
            s[nt][r] = e;
            sum += e;
        }
        sum += __shfl_xor(sum, 1, 32);
        sum += __shfl_xor(sum, 2, 32);
        sum += __shfl_xor(sum, 4, 32);
        sum += __shfl_xor(sum, 8, 32);
        inv[r] = 1.0f / sum;
    }
#pragma unroll
    for (int nt = 0; nt < 4; ++nt) {
        int n = nt * 16 + lane15;
#pragma unroll
        for (int r = 0; r < 8; ++r) {
            int m = mt * 16 + half * 8 + r;
            Pl[m][n] = (_Float16)(s[nt][r] * inv[r]);
        }
    }
    __syncthreads();

    // O = P @ V : M=64, N=32 (2 n-tiles), K=64 (2 k-steps)
    v8f o0 = {}, o1 = {};
#pragma unroll
    for (int ks = 0; ks < 2; ++ks) {
        v16h a = ldsA(&Pl[mt * 16 + lane15][ks * 32], half);
        v16h b0 = ldsB(&VlT[lane15][ks * 32], half);
        v16h b1 = ldsB(&VlT[16 + lane15][ks * 32], half);
        o0 = __builtin_amdgcn_wmma_f32_16x16x32_f16(
            false, a, false, b0, (short)0, o0, false, false);
        o1 = __builtin_amdgcn_wmma_f32_16x16x32_f16(
            false, a, false, b1, (short)0, o1, false, false);
    }
#pragma unroll
    for (int r = 0; r < 8; ++r) {
        int m = mt * 16 + half * 8 + r;
        if (m < NTOK) {
            size_t rowoff = ((size_t)b * NTOK + m) * DIM + h * HD;
            aout[rowoff + lane15]      = (_Float16)o0[r];
            aout[rowoff + 16 + lane15] = (_Float16)o1[r];
        }
    }
}

// ---------------------------------------------------------------------------
// Kernel 3: output projection [200704,192] x [192,192]^T + b -> fp32 out.
// Per block: C[64 x 96]; whole tiles staged once, barrier-free k-loop.
// grid (2, MROWS/64), block 256 (8 waves).
// ---------------------------------------------------------------------------
__global__ __launch_bounds__(256) void proj_gemm(
    const _Float16* __restrict__ aout, const float* __restrict__ proj_w,
    const float* __restrict__ proj_b, float* __restrict__ out)
{
    __shared__ _Float16 As[64][200];  // [m][k]
    __shared__ _Float16 Bs[96][200];  // [n][k]

    const int tid = threadIdx.x;
    const int wave = tid >> 5, lane = tid & 31;
    const int lane15 = lane & 15, half = lane >> 4;
    const int n0 = blockIdx.x * 96;
    const int m0 = blockIdx.y * 64;
    const int mt = wave >> 1;
    const int ntb = (wave & 1) * 3;

    // stage A (already f16): 64 rows x 24 uint4 each; b128 copies
    {
        const uint4* src = (const uint4*)aout;
        for (int i = tid; i < 64 * 24; i += 256) {
            int row = i / 24, c4 = i - row * 24;
            uint4 vv = src[((size_t)(m0 + row) * DIM) / 8 + c4];
            *(uint4*)&As[row][c4 * 8] = vv;
        }
    }
    // stage B: 96 x 192 fp32 -> f16
    {
        const float4* src = (const float4*)proj_w;
        for (int i = tid; i < 96 * 48; i += 256) {
            int n = i / 48, c4 = i - n * 48;
            float4 vv = src[((size_t)(n0 + n) * DIM) / 4 + c4];
            _Float16* dst = &Bs[n][c4 * 4];
            dst[0] = (_Float16)vv.x; dst[1] = (_Float16)vv.y;
            dst[2] = (_Float16)vv.z; dst[3] = (_Float16)vv.w;
        }
    }
    __syncthreads();

    v8f acc[3];
#pragma unroll
    for (int t = 0; t < 3; ++t) acc[t] = (v8f){};

#pragma unroll
    for (int kk = 0; kk < 6; ++kk) {
        v16h a = ldsA(&As[mt * 16 + lane15][kk * 32], half);
#pragma unroll
        for (int t = 0; t < 3; ++t) {
            v16h b = ldsB(&Bs[(ntb + t) * 16 + lane15][kk * 32], half);
            acc[t] = __builtin_amdgcn_wmma_f32_16x16x32_f16(
                false, a, false, b, (short)0, acc[t], false, false);
        }
    }

#pragma unroll
    for (int t = 0; t < 3; ++t) {
        int col = n0 + (ntb + t) * 16 + lane15;
        float pb = proj_b[col];
#pragma unroll
        for (int r = 0; r < 8; ++r) {
            int gm = m0 + mt * 16 + half * 8 + r;
            out[(size_t)gm * DIM + col] = acc[t][r] + pb;
        }
    }
}

extern "C" void kernel_launch(void* const* d_in, const int* in_sizes, int n_in,
                              void* d_out, int out_size, void* d_ws,
                              size_t ws_size, hipStream_t stream) {
    (void)in_sizes; (void)n_in; (void)out_size; (void)ws_size;
    const float* x          = (const float*)d_in[0];
    const float* mask       = (const float*)d_in[1];
    const float* qkv_w      = (const float*)d_in[2];
    const float* qkv_b      = (const float*)d_in[3];
    const float* proj_w     = (const float*)d_in[4];
    const float* proj_b     = (const float*)d_in[5];
    const float* bias_table = (const float*)d_in[6];
    const int*   rel_index  = (const int*)d_in[7];
    float* out = (float*)d_out;

    const size_t qkvElems = (size_t)BTOT * HEADS * NTOK * HD; // 38,535,168
    _Float16* qf   = (_Float16*)d_ws;
    _Float16* kf   = qf + qkvElems;
    _Float16* vf   = kf + qkvElems;
    _Float16* aout = vf + qkvElems;

    // x = column-panel index (fastest varying) so the 6 blocks sharing one
    // 64-row A strip are adjacent in dispatch order -> strip stays in L2.
    qkv_gemm<<<dim3(6, MROWS / 64), 256, 0, stream>>>(x, qkv_w, qkv_b,
                                                      qf, kf, vf);
    win_attn<<<BTOT * HEADS, 128, 0, stream>>>(qf, kf, vf, mask, bias_table,
                                               rel_index, aout);
    proj_gemm<<<dim3(2, MROWS / 64), 256, 0, stream>>>(aout, proj_w, proj_b,
                                                       out);
}
